// LSTMClassifier_72413148610970
// MI455X (gfx1250) — 8-bit matrix operand order repaired and checked by reading, compiled, not run on hardware
//
#include <hip/hip_runtime.h>
#include <hip/hip_bf16.h>


typedef int   v16i __attribute__((ext_vector_type(16)));
typedef float v8f  __attribute__((ext_vector_type(8)));

#define VOCAB   128
#define EMB     128
#define HID     256
#define NGATE   1024
#define BATCH   128
#define T_SEQ   2048
#define ROWS    16
#define NWG     (BATCH / ROWS)
#define A_STRIDE 272
#define SCL     (1.0f / 256.0f)

__device__ __forceinline__ unsigned char f32_to_e4m3(float x) {
  unsigned char s = 0;
  if (x < 0.f) { s = 0x80; x = -x; }
  if (!(x > 0.f)) return s;
  if (x >= 448.f) return s | 0x7E;
  if (x < 0.015625f) {
    int m = (int)(x * 512.f + 0.5f);
    return s | (unsigned char)m;
  }
  union { float f; unsigned u; } v; v.f = x;
  unsigned u = v.u + 0x00080000u;
  int e = (int)(u >> 23) - 120;
  unsigned m = (u >> 20) & 7u;
  if (e > 15) return s | 0x7E;
  return s | (unsigned char)((e << 3) | m);
}

__device__ __forceinline__ float fast_tanh(float x) {
#if __has_builtin(__builtin_amdgcn_tanhf)
  return __builtin_amdgcn_tanhf(x);
#elif __has_builtin(__builtin_amdgcn_tanh_f32)
  return __builtin_amdgcn_tanh_f32(x);
#else
  float e = __expf(2.f * x);
  return 1.f - 2.f / (e + 1.f);
#endif
}
__device__ __forceinline__ float fast_sigmoid(float x) {
  return 0.5f * fast_tanh(0.5f * x) + 0.5f;
}

__global__ __launch_bounds__(256) void g_table_kernel(
    const float* __restrict__ emb, const float* __restrict__ wih,
    const float* __restrict__ bih, const float* __restrict__ bhh,
    float* __restrict__ G) {
  int idx = blockIdx.x * 256 + threadIdx.x;
  int v = idx >> 10, n = idx & (NGATE - 1);
  float s = bih[n] + bhh[n];
  const float* er = emb + v * EMB;
  const float* wr = wih + n * EMB;
  #pragma unroll 8
  for (int e = 0; e < EMB; ++e) s += er[e] * wr[e];
  G[idx] = s;
}

__global__ __launch_bounds__(256) void whhq_kernel(
    const float* __restrict__ whh, unsigned char* __restrict__ wf8) {
  int i = blockIdx.x * 256 + threadIdx.x;
  wf8[i] = f32_to_e4m3(16.f * whh[i]);
}

__global__ __launch_bounds__(512, 1) void lstm_kernel(
    const int* __restrict__ x, const float* __restrict__ G,
    const unsigned char* __restrict__ wf8, float* __restrict__ h_out) {
  const int tid  = threadIdx.x;
  const int lane = tid & 31;
  const int wid  = tid >> 5;
  const int l16  = lane & 15;
  const int lg   = lane >> 4;
  const int batch0 = blockIdx.x * ROWS;

  __shared__ __align__(16) unsigned char hA[2][ROWS * A_STRIDE];
  __shared__ __align__(16) int xshare[2][ROWS];

  v16i Bf[4][2];
  #pragma unroll
  for (int g = 0; g < 4; ++g) {
    const int N = g * HID + wid * 16 + l16;
    #pragma unroll
    for (int kh = 0; kh < 2; ++kh) {
      #pragma unroll
      for (int v = 0; v < 16; ++v) {
        const int K = kh * 128 + (v >> 2) * 32 + lg * 16 + (v & 3) * 4;
        Bf[g][kh][v] = *(const int*)(wf8 + N * HID + K);
      }
    }
  }

  for (int i = tid; i < 2 * ROWS * A_STRIDE / 4; i += 512)
    ((unsigned int*)hA)[i] = 0u;
  if (tid < ROWS) xshare[0][tid] = x[(batch0 + tid) * T_SEQ];
  __syncthreads();

  float c[8], hl[8];
  #pragma unroll
  for (int r = 0; r < 8; ++r) { c[r] = 0.f; hl[r] = 0.f; }

  const int jcol = wid * 16 + l16;
  const int brow = lg * 8;

  for (int t = 0; t < T_SEQ; ++t) {
    const int p = t & 1;
    const unsigned char* Ard = hA[p];
    unsigned char*       Awr = hA[p ^ 1];

    int tok[8];
    {
      int4 ta = *(const int4*)&xshare[p][brow];
      int4 tb = *(const int4*)&xshare[p][brow + 4];
      tok[0] = ta.x; tok[1] = ta.y; tok[2] = ta.z; tok[3] = ta.w;
      tok[4] = tb.x; tok[5] = tb.y; tok[6] = tb.z; tok[7] = tb.w;
    }

    float gI[8], gF[8], gG[8], gO[8];
    #pragma unroll
    for (int r = 0; r < 8; ++r) {
      const float* Gr = G + tok[r] * NGATE + jcol;
      gI[r] = Gr[0];
      gF[r] = Gr[HID];
      gG[r] = Gr[2 * HID];
      gO[r] = Gr[3 * HID];
    }

    const int tn = (t + 1 < T_SEQ) ? t + 1 : t;
    int xn = 0;
    if (tid < ROWS) xn = x[(batch0 + tid) * T_SEQ + tn];

    v16i Af[2];
    #pragma unroll
    for (int kh = 0; kh < 2; ++kh) {
      #pragma unroll
      for (int v = 0; v < 16; v += 2) {
        const int v8 = v & 7;
        const int K = kh * 128 + (v >> 2) * 32 + lg * 16 + (v & 2) * 4;
        unsigned long long q =
            *(const unsigned long long*)(Ard + l16 * A_STRIDE + K);
        Af[kh][v]     = (int)(unsigned)q;
        Af[kh][v + 1] = (int)(unsigned)(q >> 32);
      }
    }

    v8f acc[4];
    #pragma unroll
    for (int g = 0; g < 4; ++g) {
      v8f z = {0.f, 0.f, 0.f, 0.f, 0.f, 0.f, 0.f, 0.f};
      z = __builtin_amdgcn_wmma_f32_16x16x128_fp8_fp8(Af[0], Bf[g][0],
                                                      (short)0, z, false, false);
      z = __builtin_amdgcn_wmma_f32_16x16x128_fp8_fp8(Af[1], Bf[g][1],
                                                      (short)0, z, false, false);
      acc[g] = z;
    }

    #pragma unroll
    for (int r = 0; r < 8; ++r) {
      const int b = brow + r;
      float pi = acc[0][r] * SCL + gI[r];
      float pf = acc[1][r] * SCL + gF[r];
      float pg = acc[2][r] * SCL + gG[r];
      float po = acc[3][r] * SCL + gO[r];
      float i_ = fast_sigmoid(pi);
      float f_ = fast_sigmoid(pf);
      float g_ = fast_tanh(pg);
      float o_ = fast_sigmoid(po);
      c[r] = f_ * c[r] + i_ * g_;
      float h = o_ * fast_tanh(c[r]);
      hl[r] = h;
      Awr[b * A_STRIDE + jcol] = f32_to_e4m3(16.f * h);
    }

    if (tid < ROWS) xshare[p ^ 1][tid] = xn;

    __syncthreads();
  }

  #pragma unroll
  for (int r = 0; r < 8; ++r)
    h_out[(batch0 + brow + r) * HID + jcol] = hl[r];
}

__global__ __launch_bounds__(256) void logits_kernel(
    const float* __restrict__ h, const float* __restrict__ fcW,
    const float* __restrict__ fcb, float* __restrict__ out) {
  int tid = threadIdx.x;
  int b = tid >> 1, o = tid & 1;
  float s = fcb[o];
  const float* hr = h + b * HID;
  const float* wr = fcW + o * HID;
  #pragma unroll 8
  for (int j = 0; j < HID; ++j) s += hr[j] * wr[j];
  out[b * 2 + o] = s;
}

extern "C" void kernel_launch(void* const* d_in, const int* in_sizes, int n_in,
                              void* d_out, int out_size, void* d_ws, size_t ws_size,
                              hipStream_t stream) {
  const int*   x    = (const int*)  d_in[0];
  const float* emb  = (const float*)d_in[1];
  const float* wih  = (const float*)d_in[2];
  const float* whh  = (const float*)d_in[3];
  const float* bih  = (const float*)d_in[4];
  const float* bhh  = (const float*)d_in[5];
  const float* fcW  = (const float*)d_in[6];
  const float* fcb  = (const float*)d_in[7];

  unsigned char* ws = (unsigned char*)d_ws;
  float*         G    = (float*)ws;
  unsigned char* wf8  = ws + (size_t)VOCAB * NGATE * 4;
  float*         hout = (float*)(wf8 + (size_t)NGATE * HID);

  g_table_kernel<<<(VOCAB * NGATE) / 256, 256, 0, stream>>>(emb, wih, bih, bhh, G);
  whhq_kernel<<<(NGATE * HID) / 256, 256, 0, stream>>>(whh, wf8);
  lstm_kernel<<<NWG, 512, 0, stream>>>(x, G, wf8, hout);
  logits_kernel<<<1, 256, 0, stream>>>(hout, fcW, fcb, (float*)d_out);
}
